// NTRGRU_37555194036849
// MI455X (gfx1250) — matrix operand order repaired and checked by reading, not compiled, not run on hardware
//
#include <hip/hip_runtime.h>

typedef __attribute__((ext_vector_type(16))) _Float16 v16h;
typedef __attribute__((ext_vector_type(8)))  float    v8f;

#define B_  256
#define T_  1024
#define H_  256
#define BT  16            // batch rows per workgroup (WMMA M)
#define NT  (H_/16)       // 16 N-tiles
#define KT  (H_/32)       // 8 K-slices of 32

// ---------------------------------------------------------------------------
// Pre-pass: convert fp32 weights -> f16 B-matrix fragments in d_ws.
// f16 element index: ((g*NT + nt)*KT + kt)*512 + lane*16 + s
//   B fragment order used here (contiguous k, 32x16): lane = 16*khalf + n ; slot s -> K = khalf*16 + s
// ---------------------------------------------------------------------------
__global__ void __launch_bounds__(256)
cvt_weights(const float* __restrict__ Whr,
            const float* __restrict__ Whz,
            const float* __restrict__ Whn,
            _Float16* __restrict__ ws) {
  int idx  = blockIdx.x * 256 + threadIdx.x;   // 0 .. 3*65536-1
  int s    =  idx        & 15;
  int lane = (idx >> 4)  & 31;
  int kt   = (idx >> 9)  & 7;
  int nt   = (idx >> 12) & 15;
  int g    =  idx >> 16;
  int K    = kt * 32 + (lane >> 4) * 16 + s;
  int nc   = nt * 16 + (lane & 15);
  const float* W = (g == 0) ? Whr : (g == 1) ? Whz : Whn;
  ws[idx] = (_Float16)W[K * H_ + nc];
}

// ---------------------------------------------------------------------------
// Persistent GRU scan: one workgroup = 16 batch rows, loops all T steps.
// 8 waves; wave w computes N-tiles {2w, 2w+1} of h_new via WMMA f16->f32.
// ---------------------------------------------------------------------------
__global__ void __launch_bounds__(256)
gru_scan(const float* __restrict__ h0,
         const _Float16* __restrict__ ws,
         const float* __restrict__ br,
         const float* __restrict__ bz,
         const float* __restrict__ bn,
         float* __restrict__ hfin,          // [B,H]
         float* __restrict__ y) {           // [B,T,H]
  __shared__ float h32[BT * H_];                         // fp32 state, 16 KB
  __shared__ __align__(32) _Float16 hA[KT * 32 * 16];    // A fragments, 8 KB

  const int b0    = blockIdx.x * BT;
  const int tid   = threadIdx.x;
  const int lane  = tid & 31;
  const int wave  = tid >> 5;
  const int mhalf = lane >> 4;
  const int nlo   = lane & 15;

  // load initial state h0 tile into LDS (row-major [m][c])
  for (int i = tid; i < BT * H_; i += 256)
    h32[i] = h0[(b0 + (i >> 8)) * H_ + (i & 255)];

  // loop-invariant biases for this wave's two n-tiles
  float brv[2], bzv[2], bnv[2];
  int   nc_[2];
#pragma unroll
  for (int j = 0; j < 2; ++j) {
    int nc = (wave * 2 + j) * 16 + nlo;
    nc_[j] = nc; brv[j] = br[nc]; bzv[j] = bz[nc]; bnv[j] = bn[nc];
  }

  // staging role: thread -> (kt, lane) of the A fragment layout
  const int skt = tid >> 5;
  const int sln = tid & 31;
  const int sm  = sln & 15;
  const int skh = sln >> 4;

  for (int t = 0; t < T_; ++t) {
    __syncthreads();   // h32 writes from previous step complete
    // Stage A fragments in this kernel's contiguous-k order (same as B in ws):
    //   lane = 16*khalf + m ; slot s -> K = kt*32 + khalf*16 + s
#pragma unroll
    for (int s = 0; s < 16; ++s) {
      int K = skt * 32 + skh * 16 + s;
      hA[(skt * 32 + sln) * 16 + s] = (_Float16)h32[sm * H_ + K];
    }
    __syncthreads();   // fragments visible to all waves

    v8f aR[2] = {{}, {}}, aZ[2] = {{}, {}}, aN[2] = {{}, {}};
#pragma unroll
    for (int kt = 0; kt < KT; ++kt) {
      v16h a = *(const v16h*)&hA[(kt * 32 + lane) * 16];
#pragma unroll
      for (int j = 0; j < 2; ++j) {
        int nt = wave * 2 + j;
        const _Float16* base = &ws[(nt * KT + kt) * 512 + lane * 16];
        v16h wR = *(const v16h*)(base);
        v16h wZ = *(const v16h*)(base + 1 * NT * KT * 512);
        v16h wN = *(const v16h*)(base + 2 * NT * KT * 512);
        aR[j] = __builtin_amdgcn_wmma_f32_16x16x32_f16(false, a, false, wR,
                                                       (short)0, aR[j], false, false);
        aZ[j] = __builtin_amdgcn_wmma_f32_16x16x32_f16(false, a, false, wZ,
                                                       (short)0, aZ[j], false, false);
        aN[j] = __builtin_amdgcn_wmma_f32_16x16x32_f16(false, a, false, wN,
                                                       (short)0, aN[j], false, false);
      }
    }

    // Epilogue: C/D layout lane = 16*mhalf + n, VGPR v -> M = 8*mhalf + v.
    // Each wave reads/writes only its own 32 columns of h32 -> no barrier.
#pragma unroll
    for (int j = 0; j < 2; ++j) {
      int nc = nc_[j];
#pragma unroll
      for (int v = 0; v < 8; ++v) {
        int   M  = mhalf * 8 + v;
        float ho = h32[M * H_ + nc];
        float r  = 1.0f / (1.0f + __expf(-(aR[j][v] + brv[j])));
        float z  = 1.0f / (1.0f + __expf(-(aZ[j][v] + bzv[j])));
        float nn = tanhf(r * aN[j][v] + bnv[j]);
        float hn = (1.0f - z) * nn + z * ho;
        h32[M * H_ + nc] = hn;
        y[((size_t)(b0 + M) * T_ + t) * H_ + nc] = hn;
      }
    }

    if (t == T_ - 1) {
#pragma unroll
      for (int j = 0; j < 2; ++j)
#pragma unroll
        for (int v = 0; v < 8; ++v) {
          int M = mhalf * 8 + v;
          hfin[(b0 + M) * H_ + nc_[j]] = h32[M * H_ + nc_[j]];
        }
    }
  }
}

// ---------------------------------------------------------------------------
// Inputs: 0:x(unused) 1:h0 2:Whr 3:Whz 4:Whn 5:br 6:bz 7:bn
// Output: [h_final (B*H) | y (B*T*H)] fp32
// ---------------------------------------------------------------------------
extern "C" void kernel_launch(void* const* d_in, const int* in_sizes, int n_in,
                              void* d_out, int out_size, void* d_ws, size_t ws_size,
                              hipStream_t stream) {
  const float* h0  = (const float*)d_in[1];
  const float* Whr = (const float*)d_in[2];
  const float* Whz = (const float*)d_in[3];
  const float* Whn = (const float*)d_in[4];
  const float* br  = (const float*)d_in[5];
  const float* bz  = (const float*)d_in[6];
  const float* bn  = (const float*)d_in[7];

  float* hfin = (float*)d_out;
  float* y    = hfin + B_ * H_;
  _Float16* ws = (_Float16*)d_ws;   // needs 3*H*H*2 = 384 KB

  cvt_weights<<<(3 * H_ * H_) / 256, 256, 0, stream>>>(Whr, Whz, Whn, ws);
  gru_scan<<<B_ / BT, 256, 0, stream>>>(h0, ws, br, bz, bn, hfin, y);
}
